// TwoDimensionalGRUSeq2Seq_13159779795695
// MI455X (gfx1250) — matrix operand order repaired and checked by reading, not compiled, not run on hardware
//
#include <hip/hip_runtime.h>
#include <hip/hip_bf16.h>

// ---------------------------------------------------------------------------
// 2D GRU seq2seq for MI455X (gfx1250).
// B=64, PR=PC=16, IN=64, E=128, H=256, HC=512, 3HC=1536, LAT=64, 4 dirs.
// v_wmma_f32_16x16x32_f16 for all GEMMs; wavefront over 31 anti-diagonals;
// one workgroup per (dir, cell); activation panels staged into LDS via the
// Tensor Data Mover (TENSORcnt) and consumed as ds_load_b128 A-fragments;
// weights prepacked into WMMA B-fragment lane order and streamed from L2.
// ---------------------------------------------------------------------------

typedef __attribute__((ext_vector_type(16))) _Float16 half16;
typedef __attribute__((ext_vector_type(8)))  _Float16 half8;
typedef __attribute__((ext_vector_type(8)))  float    floatx8;
typedef __attribute__((ext_vector_type(4)))  unsigned int v4u;
typedef __attribute__((ext_vector_type(8)))  int      v8i;
typedef __attribute__((ext_vector_type(4)))  int      v4i;

#define DEVI static __device__ __forceinline__

// LDS map (bytes) for the scan kernel; total 212992 (208KB) dynamic.
#define LDS_XE   0u        // f16 64x128  xe panel            (16KB)
#define LDS_HV   16384u    // f16 64x256  h_v panel / h_full  (32KB)
#define LDS_HH   49152u    // f16 64x256  h_h panel / h_full  (32KB)
#define LDS_GRZ  81920u    // f16 32x1024 fused r,z gates     (64KB)  HC f32 64x256 aliases this
#define LDS_IN   147456u   // f16 32x512  i_n                 (32KB)
#define LDS_HN   180224u   // f16 32x512  h_n                 (32KB)
#define LDS_TOT  212992u

DEVI floatx8 wmma_f16(half16 a, half16 b, floatx8 c) {
  return __builtin_amdgcn_wmma_f32_16x16x32_f16(
      false, a, false, b, (short)0, c, false, false);
}

DEVI half16 load_bfrag(const _Float16* __restrict__ Wp, int tile, int lane) {
  return *(const half16*)(Wp + (size_t)tile * 512 + lane * 16);
}

DEVI half16 make_afrag(half8 g0, half8 g1) {
  half16 a;
#pragma unroll
  for (int j = 0; j < 8; ++j) { a[j] = g0[j]; a[8 + j] = g1[j]; }
  return a;
}

DEVI half8 zero8() { half8 z = {0, 0, 0, 0, 0, 0, 0, 0}; return z; }

// A-fragment 8-half group for the gate GEMMs ([xe | h_v | h_h], K=640).
DEVI half8 a8_gate(const _Float16* XEp, const _Float16* HVp,
                   const _Float16* HHp, int m, int kb) {
  if (kb < 128) return *(const half8*)(XEp + m * 128 + kb);
  if (kb < 384) return *(const half8*)(HVp + m * 256 + (kb - 128));
  return *(const half8*)(HHp + m * 256 + (kb - 384));
}

// A-fragment group for phase 3 ([h_full | xe], K=640; h_full lives in HV/HH).
DEVI half8 a8_p3(const _Float16* XEp, const _Float16* HVp,
                 const _Float16* HHp, int m, int kb) {
  if (kb < 256) return *(const half8*)(HVp + m * 256 + kb);
  if (kb < 512) return *(const half8*)(HHp + m * 256 + (kb - 256));
  return *(const half8*)(XEp + m * 128 + (kb - 512));
}

DEVI void store_tile_h(_Float16* base, int stride, int mt, int nt,
                       floatx8 acc, int lane) {
  int n = nt * 16 + (lane & 15);
  int mb = mt * 16 + ((lane & 16) ? 8 : 0);
#pragma unroll
  for (int v = 0; v < 8; ++v) base[(mb + v) * stride + n] = (_Float16)acc[v];
}

// ---------------------------------------------------------------------------
// Tensor Data Mover: async 2-D tile load global -> LDS (rows packed
// back-to-back, row length == tile_dim0). Descriptor per ISA ch.8.
// This toolchain's builtin takes 6 args (extra int32x8 group before cpol).
// ---------------------------------------------------------------------------
#if __has_builtin(__builtin_amdgcn_tensor_load_to_lds) && \
    __has_builtin(__builtin_amdgcn_s_wait_tensorcnt)
#define HAVE_TDM 1
DEVI void tdm_load_2d(unsigned lds_off, const void* gptr,
                      unsigned tile_d0, unsigned tile_d1,
                      unsigned long long stride0_elems) {
  unsigned long long ga = (unsigned long long)gptr;
  v4u g0;
  g0[0] = 1u;                                   // count=1, user descriptor
  g0[1] = lds_off;                              // lds_addr (bytes)
  g0[2] = (unsigned)(ga & 0xffffffffull);       // global_addr[31:0]
  g0[3] = (unsigned)((ga >> 32) & 0x1ffffffull) // global_addr[56:32]
        | (2u << 30);                           // type = 2 ("image")
  v8i g1;
  g1[0] = (int)(1u << 16);                      // data_size = 1 (2 bytes)
  g1[1] = (int)((tile_d0 & 0xffffu) << 16);     // tensor_dim0 lo16 at bit48
  g1[2] = (int)((tile_d0 >> 16) | ((tile_d1 & 0xffffu) << 16)); // td0hi|td1lo
  g1[3] = (int)((tile_d1 >> 16) | ((tile_d0 & 0xffffu) << 16)); // td1hi|tile_dim0
  g1[4] = (int)(tile_d1 & 0xffffu);             // tile_dim1 (tile_dim2 = 0)
  g1[5] = (int)(stride0_elems & 0xffffffffull); // tensor_dim0_stride[31:0]
  g1[6] = (int)((stride0_elems >> 32) & 0xffffull); // stride0[47:32]
  g1[7] = 0;
  v4i z4 = {0, 0, 0, 0};
  v8i z8 = {0, 0, 0, 0, 0, 0, 0, 0};
  __builtin_amdgcn_tensor_load_to_lds(g0, g1, z4, z4, z8, 0);
}
#else
#define HAVE_TDM 0
#endif

// ---------------------------------------------------------------------------
// Weight prepack: logical W[k][n] -> WMMA B-fragment lane order:
//   tile (kt=k/32, nt=n/16); lane = ((k&15)>=8?16:0)+(n&15); j = (k&7)+((k&16)?8:0).
// ---------------------------------------------------------------------------
DEVI size_t packed_off(int tile, int k, int n) {
  int lane = (((k & 15) >= 8) ? 16 : 0) + (n & 15);
  return ((size_t)tile * 32 + lane) * 16 + (k & 7) + ((k & 16) ? 8 : 0);
}

__global__ void pack_w1_kernel(const float* __restrict__ w_ih,
                               const float* __restrict__ w_hh,
                               _Float16* __restrict__ W1P) {
  size_t idx = (size_t)blockIdx.x * 256 + threadIdx.x;
  if (idx >= (size_t)4 * 640 * 1024) return;
  int d = (int)(idx / (640 * 1024));
  int rem = (int)(idx % (640 * 1024));
  int k = rem / 1024, n = rem % 1024;
  float v = (k < 128)
      ? w_ih[(size_t)d * 1536 * 128 + (size_t)n * 128 + k]
      : w_hh[(size_t)d * 1536 * 512 + (size_t)n * 512 + (k - 128)];
  int tile = (d * 20 + (k >> 5)) * 64 + (n >> 4);
  W1P[packed_off(tile, k, n)] = (_Float16)v;
}

__global__ void pack_wn_ih_kernel(const float* __restrict__ w_ih,
                                  _Float16* __restrict__ WNIP) {
  size_t idx = (size_t)blockIdx.x * 256 + threadIdx.x;
  if (idx >= (size_t)4 * 128 * 512) return;
  int d = (int)(idx / (128 * 512));
  int rem = (int)(idx % (128 * 512));
  int k = rem / 512, n = rem % 512;
  float v = w_ih[(size_t)d * 1536 * 128 + (size_t)(1024 + n) * 128 + k];
  int tile = (d * 4 + (k >> 5)) * 32 + (n >> 4);
  WNIP[packed_off(tile, k, n)] = (_Float16)v;
}

__global__ void pack_wn_hh_kernel(const float* __restrict__ w_hh,
                                  _Float16* __restrict__ WNHP) {
  size_t idx = (size_t)blockIdx.x * 256 + threadIdx.x;
  if (idx >= (size_t)4 * 512 * 512) return;
  int d = (int)(idx / (512 * 512));
  int rem = (int)(idx % (512 * 512));
  int k = rem / 512, n = rem % 512;
  float v = w_hh[(size_t)d * 1536 * 512 + (size_t)(1024 + n) * 512 + k];
  int tile = (d * 16 + (k >> 5)) * 32 + (n >> 4);
  WNHP[packed_off(tile, k, n)] = (_Float16)v;
}

__global__ void pack_w2_kernel(const float* __restrict__ w_comp,
                               const float* __restrict__ w_res,
                               _Float16* __restrict__ W2P) {
  size_t idx = (size_t)blockIdx.x * 256 + threadIdx.x;
  if (idx >= (size_t)4 * 640 * 256) return;
  int d = (int)(idx / (640 * 256));
  int rem = (int)(idx % (640 * 256));
  int k = rem / 256, n = rem % 256;
  float v = (k < 512)
      ? w_comp[(size_t)d * 256 * 512 + (size_t)n * 512 + k]
      : w_res[(size_t)d * 256 * 128 + (size_t)n * 128 + (k - 512)];
  int tile = (d * 20 + (k >> 5)) * 16 + (n >> 4);
  W2P[packed_off(tile, k, n)] = (_Float16)v;
}

__global__ void pack_bias2_kernel(const float* __restrict__ b_comp,
                                  const float* __restrict__ b_res,
                                  float* __restrict__ bias2) {
  int idx = blockIdx.x * 256 + threadIdx.x;
  if (idx < 1024) bias2[idx] = b_comp[idx] + b_res[idx];
}

// ---------------------------------------------------------------------------
// Embedding: xe(16384x128) = x(16384x64) @ w_emb.T + b_emb  -> f16
// ---------------------------------------------------------------------------
__global__ __launch_bounds__(256) void embed_kernel(
    const float* __restrict__ x, const float* __restrict__ w_emb,
    const float* __restrict__ b_emb, _Float16* __restrict__ xe16) {
  int mt = blockIdx.x;
  int lane = threadIdx.x & 31, wv = threadIdx.x >> 5;
  int m = mt * 16 + (lane & 15);
  int n = wv * 16 + (lane & 15);
  floatx8 acc = {};
#pragma unroll
  for (int kt = 0; kt < 2; ++kt) {
    int k0 = kt * 32;
    half16 a, b;
#pragma unroll
    for (int j = 0; j < 16; ++j) {
      int g = j >> 1, h = j & 1;
      int ka = k0 + ((g >= 4) ? 16 : 0) + ((lane & 16) ? 8 : 0) + ((g & 3) << 1) + h;
      a[j] = (_Float16)x[(size_t)m * 64 + ka];
      int kb = k0 + ((j >= 8) ? 16 : 0) + ((lane & 16) ? 8 : 0) + (j & 7);
      b[j] = (_Float16)w_emb[(size_t)n * 64 + kb];
    }
    acc = wmma_f16(a, b, acc);
  }
  int mb = mt * 16 + ((lane & 16) ? 8 : 0);
  float be = b_emb[n];
#pragma unroll
  for (int v = 0; v < 8; ++v)
    xe16[(size_t)(mb + v) * 128 + n] = (_Float16)(acc[v] + be);
}

// ---------------------------------------------------------------------------
// Per-diagonal 2D-GRU cell kernel. One workgroup (8 waves) per (dir, cell).
// Stage xe/h_v/h_h panels to LDS via TDM; gate GEMMs over two batch halves;
// h_full written in place over HV/HH; comp GEMM + LayerNorm; f16 grid out.
// ---------------------------------------------------------------------------
__global__ __launch_bounds__(256) void scan_diag_kernel(
    const _Float16* __restrict__ xe16, _Float16* __restrict__ outh16,
    const _Float16* __restrict__ W1P, const _Float16* __restrict__ WNIP,
    const _Float16* __restrict__ WNHP, const _Float16* __restrict__ W2P,
    const float* __restrict__ bias2, const float* __restrict__ b_ih,
    const float* __restrict__ b_hh, const float* __restrict__ ln_g,
    const float* __restrict__ ln_b, int diag) {
  extern __shared__ char smem[];
  _Float16* XEp = (_Float16*)(smem + LDS_XE);
  _Float16* HVp = (_Float16*)(smem + LDS_HV);
  _Float16* HHp = (_Float16*)(smem + LDS_HH);
  _Float16* GRZ = (_Float16*)(smem + LDS_GRZ);
  _Float16* INb = (_Float16*)(smem + LDS_IN);
  _Float16* HNb = (_Float16*)(smem + LDS_HN);
  float*    HC  = (float*)(smem + LDS_GRZ);   // f32 64x256, aliases GRZ

  int rmin = (diag - 15 > 0) ? diag - 15 : 0;
  int rmax = (diag < 15) ? diag : 15;
  int ncells = rmax - rmin + 1;
  int d = blockIdx.x / ncells;
  int ci = blockIdx.x - d * ncells;
  int r = rmin + ci, c = diag - r;
  bool fr = (d >= 2), fc = (d & 1);
  int rp = fr ? 15 - r : r;
  int cp = fc ? 15 - c : c;
  int rv = rp + (fr ? 1 : -1);
  int cv = cp + (fc ? 1 : -1);
  bool vv = (r > 0), vh = (c > 0);

  int tid = threadIdx.x, lane = tid & 31, wv = tid >> 5;
  int koff = (lane & 16) ? 8 : 0;
  int ml = lane & 15;

  // ---- Stage activation panels into LDS ----
#if HAVE_TDM
  if (wv == 0) {
    tdm_load_2d(LDS_XE, xe16 + (size_t)(rp * 16 + cp) * 128, 128, 64, 32768ull);
    if (vv)
      tdm_load_2d(LDS_HV,
                  outh16 + ((size_t)((d * 64) * 16 + rv) * 16 + cp) * 256,
                  256, 64, 65536ull);
    if (vh)
      tdm_load_2d(LDS_HH,
                  outh16 + ((size_t)((d * 64) * 16 + rp) * 16 + cv) * 256,
                  256, 64, 65536ull);
    __builtin_amdgcn_s_wait_tensorcnt(0);
  }
  if (!vv) for (int i = tid; i < 2048; i += 256) ((half8*)HVp)[i] = zero8();
  if (!vh) for (int i = tid; i < 2048; i += 256) ((half8*)HHp)[i] = zero8();
#else
  for (int i = tid; i < 1024; i += 256) {       // xe: 64 rows x 16 half8
    int m = i >> 4, o = (i & 15) * 8;
    ((half8*)XEp)[i] =
        *(const half8*)(xe16 + ((size_t)(m * 256 + rp * 16 + cp)) * 128 + o);
  }
  for (int i = tid; i < 2048; i += 256) {       // h_v: 64 rows x 32 half8
    int m = i >> 5, o = (i & 31) * 8;
    ((half8*)HVp)[i] = vv
        ? *(const half8*)(outh16 +
              ((size_t)((d * 64 + m) * 16 + rv) * 16 + cp) * 256 + o)
        : zero8();
  }
  for (int i = tid; i < 2048; i += 256) {       // h_h: 64 rows x 32 half8
    int m = i >> 5, o = (i & 31) * 8;
    ((half8*)HHp)[i] = vh
        ? *(const half8*)(outh16 +
              ((size_t)((d * 64 + m) * 16 + rp) * 16 + cv) * 256 + o)
        : zero8();
  }
#endif
  __syncthreads();

  // ---- Gate stage over two batch halves (rows m0..m0+31) ----
  const float* bi = b_ih + d * 1536;
  const float* bh = b_hh + d * 1536;
  for (int half = 0; half < 2; ++half) {
    int m0 = half * 32;
    // GRZ = [xe|h_v|h_h](32x640) @ W1(640x1024)
    for (int t = wv; t < 128; t += 8) {
      int mt = t >> 6, nt = t & 63;
      int m = m0 + mt * 16 + ml;
      floatx8 acc = {};
      for (int kt = 0; kt < 20; ++kt) {
        int k0 = kt * 32;
        half8 g0 = a8_gate(XEp, HVp, HHp, m, k0 + koff);
        half8 g1 = a8_gate(XEp, HVp, HHp, m, k0 + 16 + koff);
        if (kt + 1 < 20)
          __builtin_prefetch(W1P + (size_t)((d * 20 + kt + 1) * 64 + nt) * 512 +
                                 lane * 16, 0, 1);
        half16 b = load_bfrag(W1P, (d * 20 + kt) * 64 + nt, lane);
        acc = wmma_f16(make_afrag(g0, g1), b, acc);
      }
      store_tile_h(GRZ, 1024, mt, nt, acc, lane);
    }
    // INb = xe(32x128) @ Wn_ih(128x512)
    for (int t = wv; t < 64; t += 8) {
      int mt = t >> 5, nt = t & 31;
      int m = m0 + mt * 16 + ml;
      floatx8 acc = {};
      for (int kt = 0; kt < 4; ++kt) {
        int k0 = kt * 32;
        half8 g0 = a8_gate(XEp, HVp, HHp, m, k0 + koff);
        half8 g1 = a8_gate(XEp, HVp, HHp, m, k0 + 16 + koff);
        half16 b = load_bfrag(WNIP, (d * 4 + kt) * 32 + nt, lane);
        acc = wmma_f16(make_afrag(g0, g1), b, acc);
      }
      store_tile_h(INb, 512, mt, nt, acc, lane);
    }
    // HNb = hprev(32x512) @ Wn_hh(512x512)
    for (int t = wv; t < 64; t += 8) {
      int mt = t >> 5, nt = t & 31;
      int m = m0 + mt * 16 + ml;
      floatx8 acc = {};
      for (int kt = 0; kt < 16; ++kt) {
        int k0 = 128 + kt * 32;
        half8 g0 = a8_gate(XEp, HVp, HHp, m, k0 + koff);
        half8 g1 = a8_gate(XEp, HVp, HHp, m, k0 + 16 + koff);
        half16 b = load_bfrag(WNHP, (d * 16 + kt) * 32 + nt, lane);
        acc = wmma_f16(make_afrag(g0, g1), b, acc);
      }
      store_tile_h(HNb, 512, mt, nt, acc, lane);
    }
    __syncthreads();

    // Elementwise GRU -> h_full written over HV/HH panels
    for (int i = tid; i < 32 * 512; i += 256) {
      int lm = i >> 9, n = i & 511;
      int m = m0 + lm;
      float g_r = (float)GRZ[lm * 1024 + n] + bi[n] + bh[n];
      float g_z = (float)GRZ[lm * 1024 + 512 + n] + bi[512 + n] + bh[512 + n];
      float i_n = (float)INb[lm * 512 + n] + bi[1024 + n];
      float h_n = (float)HNb[lm * 512 + n] + bh[1024 + n];
      float rr = 1.f / (1.f + __expf(-g_r));
      float zz = 1.f / (1.f + __expf(-g_z));
      float nn = tanhf(i_n + rr * h_n);
      _Float16* slot = (n < 256) ? (HVp + m * 256 + n) : (HHp + m * 256 + (n - 256));
      float hp = (float)*slot;
      *slot = (_Float16)((1.f - zz) * nn + zz * hp);
    }
    __syncthreads();
  }

  // ---- Phase 3: HC = [h_full | xe](64x640) @ W2(640x256) + (b_comp+b_res) ----
  for (int t = wv; t < 64; t += 8) {
    int mt = t >> 4, nt = t & 15;
    int m = mt * 16 + ml;
    floatx8 acc = {};
    for (int kt = 0; kt < 20; ++kt) {
      int k0 = kt * 32;
      half8 g0 = a8_p3(XEp, HVp, HHp, m, k0 + koff);
      half8 g1 = a8_p3(XEp, HVp, HHp, m, k0 + 16 + koff);
      half16 b = load_bfrag(W2P, (d * 20 + kt) * 16 + nt, lane);
      acc = wmma_f16(make_afrag(g0, g1), b, acc);
    }
    int n = nt * 16 + (lane & 15);
    int mb = mt * 16 + ((lane & 16) ? 8 : 0);
    float bb = bias2[d * 256 + n];
#pragma unroll
    for (int v = 0; v < 8; ++v) HC[(mb + v) * 256 + n] = acc[v] + bb;
  }
  __syncthreads();

  // ---- Phase 4: LayerNorm over 256, write f16 grid output ----
  {
    int m = tid >> 2, q = tid & 3;
    float s = 0.f, s2 = 0.f;
    for (int j = 0; j < 64; ++j) {
      float v = HC[m * 256 + q * 64 + j];
      s += v; s2 += v * v;
    }
    s += __shfl_xor(s, 1, 4);   s += __shfl_xor(s, 2, 4);
    s2 += __shfl_xor(s2, 1, 4); s2 += __shfl_xor(s2, 2, 4);
    float mean = s * (1.f / 256.f);
    float var = s2 * (1.f / 256.f) - mean * mean;
    float rstd = rsqrtf(var + 1e-5f);
    const float* g = ln_g + d * 256;
    const float* bb = ln_b + d * 256;
    _Float16* po = outh16 + ((size_t)((d * 64 + m) * 16 + rp) * 16 + cp) * 256;
    for (int j = 0; j < 64; ++j) {
      int n = q * 64 + j;
      po[n] = (_Float16)((HC[m * 256 + n] - mean) * rstd * g[n] + bb[n]);
    }
  }
}

// ---------------------------------------------------------------------------
// Final: direction-reduce boundary rows/cols -> latent(64) -> logvar/mean.
// ---------------------------------------------------------------------------
__global__ __launch_bounds__(256) void final_kernel(
    const _Float16* __restrict__ outh16,
    const float* __restrict__ w_cr_row, const float* __restrict__ b_cr_row,
    const float* __restrict__ w_cr_col, const float* __restrict__ b_cr_col,
    const float* __restrict__ w_rowc, const float* __restrict__ b_rowc,
    const float* __restrict__ w_colc, const float* __restrict__ b_colc,
    const float* __restrict__ w_lv, const float* __restrict__ b_lv,
    const float* __restrict__ w_mu, const float* __restrict__ b_mu,
    float* __restrict__ out) {
  __shared__ float frr[4096];
  __shared__ float fcr[4096];
  __shared__ float lat[64];
  int b = blockIdx.x, tid = threadIdx.x;

  for (int i = tid; i < 4096; i += 256) {
    int cc = i >> 8, h = i & 255;
    float sr = b_cr_row[0], sc = b_cr_col[0];
#pragma unroll
    for (int d = 0; d < 4; ++d) {
      bool fr = (d >= 2), fc = (d & 1);
      int rowsel = fr ? 0 : 15, colsel = fc ? 0 : 15;
      sr += w_cr_row[d] *
            (float)outh16[((size_t)((d * 64 + b) * 16 + rowsel) * 16 + cc) * 256 + h];
      sc += w_cr_col[d] *
            (float)outh16[((size_t)((d * 64 + b) * 16 + cc) * 16 + colsel) * 256 + h];
    }
    frr[i] = sr;
    fcr[i] = sc;
  }
  __syncthreads();
  {
    int l = tid >> 2, q = tid & 3;
    const float* W = (l < 32) ? (w_rowc + (size_t)l * 4096)
                              : (w_colc + (size_t)(l - 32) * 4096);
    const float* src = (l < 32) ? frr : fcr;
    float s = 0.f;
    for (int i = q; i < 4096; i += 4) s += src[i] * W[i];
    s += __shfl_xor(s, 1, 4);
    s += __shfl_xor(s, 2, 4);
    if (q == 0) lat[l] = s + ((l < 32) ? b_rowc[l] : b_colc[l - 32]);
  }
  __syncthreads();
  if (tid < 128) {
    int j = tid & 63;
    bool is_mu = (tid >= 64);
    const float* W = is_mu ? w_mu : w_lv;
    const float* bb = is_mu ? b_mu : b_lv;
    float s = bb[j];
    for (int l = 0; l < 64; ++l) s += lat[l] * W[j * 64 + l];
    out[(is_mu ? 4096 : 0) + b * 64 + j] = s;
  }
}

// ---------------------------------------------------------------------------
extern "C" void kernel_launch(void* const* d_in, const int* in_sizes, int n_in,
                              void* d_out, int out_size, void* d_ws, size_t ws_size,
                              hipStream_t stream) {
  (void)in_sizes; (void)n_in; (void)out_size; (void)ws_size;
  const float* x        = (const float*)d_in[0];
  const float* w_emb    = (const float*)d_in[1];
  const float* b_emb    = (const float*)d_in[2];
  const float* w_ih     = (const float*)d_in[3];
  const float* b_ih     = (const float*)d_in[4];
  const float* w_hh     = (const float*)d_in[5];
  const float* b_hh     = (const float*)d_in[6];
  const float* w_comp   = (const float*)d_in[7];
  const float* b_comp   = (const float*)d_in[8];
  const float* w_res    = (const float*)d_in[9];
  const float* b_res    = (const float*)d_in[10];
  const float* ln_g     = (const float*)d_in[11];
  const float* ln_b     = (const float*)d_in[12];
  const float* w_cr_row = (const float*)d_in[13];
  const float* b_cr_row = (const float*)d_in[14];
  const float* w_cr_col = (const float*)d_in[15];
  const float* b_cr_col = (const float*)d_in[16];
  const float* w_rowc   = (const float*)d_in[17];
  const float* b_rowc   = (const float*)d_in[18];
  const float* w_colc   = (const float*)d_in[19];
  const float* b_colc   = (const float*)d_in[20];
  const float* w_lv     = (const float*)d_in[21];
  const float* b_lv     = (const float*)d_in[22];
  const float* w_mu     = (const float*)d_in[23];
  const float* b_mu     = (const float*)d_in[24];

  char* ws = (char*)d_ws;
  _Float16* XE16 = (_Float16*)(ws + 0);                 //  4,194,304 B
  _Float16* W1P  = (_Float16*)(ws + 4194304);           //  5,242,880 B
  _Float16* WNIP = (_Float16*)(ws + 9437184);           //    524,288 B
  _Float16* WNHP = (_Float16*)(ws + 9961472);           //  2,097,152 B
  _Float16* W2P  = (_Float16*)(ws + 12058624);          //  1,310,720 B
  float*    BIAS2= (float*)   (ws + 13369344);          //      4,096 B
  _Float16* OUTH = (_Float16*)(ws + 13373440);          // 33,554,432 B

  pack_w1_kernel<<<(4 * 640 * 1024 + 255) / 256, 256, 0, stream>>>(w_ih, w_hh, W1P);
  pack_wn_ih_kernel<<<(4 * 128 * 512 + 255) / 256, 256, 0, stream>>>(w_ih, WNIP);
  pack_wn_hh_kernel<<<(4 * 512 * 512 + 255) / 256, 256, 0, stream>>>(w_hh, WNHP);
  pack_w2_kernel<<<(4 * 640 * 256 + 255) / 256, 256, 0, stream>>>(w_comp, w_res, W2P);
  pack_bias2_kernel<<<4, 256, 0, stream>>>(b_comp, b_res, BIAS2);

  embed_kernel<<<1024, 256, 0, stream>>>(x, w_emb, b_emb, XE16);

  for (int diag = 0; diag <= 30; ++diag) {
    int rmin = (diag - 15 > 0) ? diag - 15 : 0;
    int rmax = (diag < 15) ? diag : 15;
    int nc = rmax - rmin + 1;
    scan_diag_kernel<<<4 * nc, 256, LDS_TOT, stream>>>(
        XE16, OUTH, W1P, WNIP, WNHP, W2P, BIAS2, b_ih, b_hh, ln_g, ln_b, diag);
  }

  final_kernel<<<64, 256, 0, stream>>>(OUTH, w_cr_row, b_cr_row, w_cr_col,
                                       b_cr_col, w_rowc, b_rowc, w_colc, b_colc,
                                       w_lv, b_lv, w_mu, b_mu, (float*)d_out);
}
